// OPTAttention_27204322853720
// MI455X (gfx1250) — matrix operand order repaired and checked by reading, not compiled, not run on hardware
//
#include <hip/hip_runtime.h>
#include <hip/hip_bf16.h>

typedef __attribute__((ext_vector_type(16))) __bf16 v16bf;
typedef __attribute__((ext_vector_type(8)))  __bf16 v8bf;
typedef __attribute__((ext_vector_type(8)))  float  v8f;

#define WMMA_BF16(A_, B_, C_) \
  __builtin_amdgcn_wmma_f32_16x16x32_bf16(false, (A_), false, (B_), (short)0, (C_), false, false)

// Problem sizes (compile-time)
#define BB 2
#define SS 2048
#define EE 2048
#define HH 32
#define DD 64
#define MM (BB * SS)   // 4096 rows
#define N_QKV (3 * EE) // 6144

// ---------------------------------------------------------------------------
// Async-to-LDS helpers (gfx1250).  All kernels using these place every LDS
// object in dynamic shared memory (no static __shared__), so dynamic LDS
// starts at byte offset 0 and the destinations are plain byte offsets.
// Each instruction moves 16B per lane and bumps the wave's ASYNCcnt by 1.
// ---------------------------------------------------------------------------
__device__ __forceinline__ void async_b128(unsigned int lds_byte_off, const void* gptr) {
  asm volatile("global_load_async_to_lds_b128 %0, %1, off"
               :: "v"(lds_byte_off), "v"(gptr) : "memory");
}
__device__ __forceinline__ void wait_async_le6() {
  asm volatile("s_wait_asynccnt 0x6" ::: "memory");
}
__device__ __forceinline__ void wait_async_le2() {
  asm volatile("s_wait_asynccnt 0x2" ::: "memory");
}
__device__ __forceinline__ void wait_async_0() {
  asm volatile("s_wait_asynccnt 0x0" ::: "memory");
}

// ---------------------------------------------------------------------------
// GEMM tile geometry: K-step 64, LDS rows padded to 144 B (36 dwords) so the
// 16-lane b128 fragment reads cover all 64 banks conflict-free.
// ---------------------------------------------------------------------------
#define KSTEP 64
#define LDT   144                               // bytes per LDS tile row
#define A_TILE_BYTES (128 * LDT)                // 18432
#define B_TILE_BYTES (64 * LDT)                 //  9216
#define ABASE(buf) ((unsigned)((buf) * A_TILE_BYTES))
#define BBASE(buf) ((unsigned)(2 * A_TILE_BYTES + (buf) * B_TILE_BYTES))
#define GEMM_SMEM  (2 * A_TILE_BYTES + 2 * B_TILE_BYTES)  // 55296 B

// Stage a ROWS x 64 bf16 tile: 8 x 16B units per row, 256 threads.
template <int ROWS, int UNITS_PER_THREAD>
__device__ __forceinline__ void load_tile_async(unsigned base, const __bf16* gsrc,
                                                int ldg, int tid) {
#pragma unroll
  for (int i = 0; i < UNITS_PER_THREAD; ++i) {
    const int u   = tid + i * 256;
    const int row = u >> 3;
    const int cu  = u & 7;
    async_b128(base + (unsigned)(row * LDT + cu * 16),
               gsrc + (size_t)row * ldg + cu * 8);
  }
}

// ---------------------------------------------------------------------------
// Elementwise fp32 -> bf16
// ---------------------------------------------------------------------------
__global__ __launch_bounds__(256) void cvt_bf16(const float* __restrict__ in,
                                                __bf16* __restrict__ out, int n) {
  int i = blockIdx.x * blockDim.x + threadIdx.x;
  if (i < n) out[i] = (__bf16)in[i];
}

// Tiled transpose + convert: in [K,N] fp32 row-major -> out [N,K] bf16.
__global__ __launch_bounds__(256) void cvt_t_bf16(const float* __restrict__ in,
                                                  __bf16* __restrict__ out,
                                                  int K, int N) {
  __shared__ float tile[32][33];
  const int k0 = blockIdx.x * 32;
  const int n0 = blockIdx.y * 32;
  const int tx = threadIdx.x & 31;
  const int ty = threadIdx.x >> 5;  // 0..7
#pragma unroll
  for (int r = ty; r < 32; r += 8)
    tile[r][tx] = in[(size_t)(k0 + r) * N + (n0 + tx)];
  __syncthreads();
#pragma unroll
  for (int r = ty; r < 32; r += 8)
    out[(size_t)(n0 + r) * K + (k0 + tx)] = (__bf16)tile[tx][r];
}

// ---------------------------------------------------------------------------
// QKV projection GEMM: C[4096,6144] = A[4096,2048] * BT[6144,2048]^T + bias.
// Double-buffered async-to-LDS staging; wave -> 16x64 tile; block -> 128x64.
// All fragments loaded before the WMMA chain to overlap DS latency.
// ---------------------------------------------------------------------------
__global__ __launch_bounds__(256) void qkv_gemm(const __bf16* __restrict__ A,
                                                const __bf16* __restrict__ BT,
                                                const float* __restrict__ bias,
                                                __bf16* __restrict__ Qo,
                                                __bf16* __restrict__ Ko,
                                                __bf16* __restrict__ Vo) {
  extern __shared__ char smem[];
  const int tid  = threadIdx.x;
  const int lane = tid & 31;
  const int wid  = tid >> 5;
  const int half = lane >> 4;
  const int l15  = lane & 15;
  const int m0b = blockIdx.x * 128;
  const int n0  = blockIdx.y * 64;
  const __bf16* Ablk = A + (size_t)m0b * EE;
  const __bf16* Bblk = BT + (size_t)n0 * EE;

  v8f acc[4] = {};

  auto compute_tile = [&](int buf) {
    const char* abase = smem + ABASE(buf) + (wid * 16 + l15) * LDT + half * 32;
    const char* bbase = smem + BBASE(buf) + l15 * LDT + half * 32;
    v16bf a[2], bf[2][4];
#pragma unroll
    for (int ks = 0; ks < 2; ++ks) {
      v8bf lo = *(const v8bf*)(abase + ks * 64);
      v8bf hi = *(const v8bf*)(abase + ks * 64 + 16);
#pragma unroll
      for (int i = 0; i < 8; ++i) { a[ks][i] = lo[i]; a[ks][8 + i] = hi[i]; }
#pragma unroll
      for (int t = 0; t < 4; ++t)
        bf[ks][t] = *(const v16bf*)(bbase + t * 16 * LDT + ks * 64);
    }
#pragma unroll
    for (int ks = 0; ks < 2; ++ks)
#pragma unroll
      for (int t = 0; t < 4; ++t)
        acc[t] = WMMA_BF16(a[ks], bf[ks][t], acc[t]);
  };

  // Prologue: stage tile 0.
  load_tile_async<128, 4>(ABASE(0), Ablk, EE, tid);
  load_tile_async<64, 2>(BBASE(0), Bblk, EE, tid);

  for (int kk = 0; kk < EE - KSTEP; kk += KSTEP) {
    const int buf = (kk >> 6) & 1;
    load_tile_async<128, 4>(ABASE(buf ^ 1), Ablk + kk + KSTEP, EE, tid);
    load_tile_async<64, 2>(BBASE(buf ^ 1), Bblk + kk + KSTEP, EE, tid);
    wait_async_le6();   // this tile's 6 older ops complete (in-order)
    __syncthreads();
    compute_tile(buf);
    __syncthreads();    // reads done before this buffer is refilled
  }
  wait_async_0();
  __syncthreads();
  compute_tile(((EE - KSTEP) >> 6) & 1);

  const float qscale = 0.125f;  // D^-0.5, folded into Q
#pragma unroll
  for (int t = 0; t < 4; ++t) {
    const int col = n0 + t * 16 + l15;
    const float bs = bias[col];
    const int sect = col >> 11;  // 0=Q, 1=K, 2=V
    const int e = col & 2047;
    const int h = e >> 6;
    const int d = e & 63;
#pragma unroll
    for (int v = 0; v < 8; ++v) {
      const int row = m0b + wid * 16 + v + 8 * half;
      const int b = row >> 11;
      const int s = row & 2047;
      const float val = acc[t][v] + bs;
      if (sect == 0) {
        Qo[(((size_t)b * HH + h) * SS + s) * DD + d] = (__bf16)(val * qscale);
      } else if (sect == 1) {
        Ko[(((size_t)b * HH + h) * SS + s) * DD + d] = (__bf16)val;
      } else {
        Vo[(((size_t)b * HH + h) * DD + d) * SS + s] = (__bf16)val;  // transposed
      }
    }
  }
}

// ---------------------------------------------------------------------------
// Causal flash attention with block-shared, double-buffered async K/V staging.
// One block per (b*H + h, q-block of 128 rows); each wave owns 16 query rows.
// All waves run the block-uniform kb loop (whole-wave guard keeps EXEC all-1s
// inside WMMA); K/V tiles are staged once per block instead of once per wave.
// LDS (dynamic, byte offsets):
//   Kbuf[2]: 32 keys x 64d bf16, row stride 144  ->  2 * 4608
//   Vbuf[2]: 64 d  x 32 keys bf16, row stride 80 ->  2 * 5120   @ 9216
//   P[8]   : 16 x 32 bf16 per wave, row stride 80 -> 8 * 1280   @ 19456
// ---------------------------------------------------------------------------
#define FKB(buf) ((unsigned)((buf) * 4608))
#define FVB(buf) ((unsigned)(9216 + (buf) * 5120))
#define FPB 19456u
#define FLASH_SMEM 29696

__global__ __launch_bounds__(256) void flash_attn(const __bf16* __restrict__ Q,
                                                  const __bf16* __restrict__ K,
                                                  const __bf16* __restrict__ V,
                                                  __bf16* __restrict__ O) {
  extern __shared__ char smem[];
  const int tid  = threadIdx.x;
  const int lane = tid & 31;
  const int wid  = tid >> 5;
  const int half = lane >> 4;
  const int l15  = lane & 15;
  const int bh = blockIdx.x;
  const int b  = bh >> 5;
  const int h  = bh & 31;
  const int q0b = blockIdx.y * 128;
  const int q0  = q0b + wid * 16;

  const __bf16* Qh = Q + (size_t)bh * SS * DD;
  const __bf16* Kh = K + (size_t)bh * SS * DD;
  const __bf16* Vh = V + (size_t)bh * DD * SS;
  char* pl = smem + FPB + wid * 1280;

  // Hoist Q A-fragments (d split into two K=32 chunks)
  const __bf16* qrowp = Qh + (size_t)(q0 + l15) * DD + half * 16;
  v16bf aq[2];
#pragma unroll
  for (int c = 0; c < 2; ++c) {
    v8bf lo = *(const v8bf*)(qrowp + c * 32);
    v8bf hi = *(const v8bf*)(qrowp + c * 32 + 8);
#pragma unroll
    for (int i = 0; i < 8; ++i) { aq[c][i] = lo[i]; aq[c][8 + i] = hi[i]; }
  }

  v8f acc[4] = {};
  float mi[8], li[8];
#pragma unroll
  for (int v = 0; v < 8; ++v) { mi[v] = -3.0e38f; li[v] = 0.0f; }

  // Cooperative staging of one K/V block (keys [kb, kb+32)): 2 async/thread.
  auto stage_kv = [&](int buf, int kb) {
    { const int row = tid >> 3, cu = tid & 7;   // K: 32 rows x 128B
      async_b128(FKB(buf) + (unsigned)(row * 144 + cu * 16),
                 Kh + (size_t)(kb + row) * DD + cu * 8); }
    { const int row = tid >> 2, cu = tid & 3;   // V^T: 64 rows x 64B
      async_b128(FVB(buf) + (unsigned)(row * 80 + cu * 16),
                 Vh + (size_t)row * SS + kb + cu * 8); }
  };

  auto fa_step = [&](int buf, int kb) {
    // --- scores: two 16x16 tiles over keys [kb, kb+32) from LDS ---
    float s2[2][8];
#pragma unroll
    for (int j = 0; j < 2; ++j) {
      const char* krow = smem + FKB(buf) + (j * 16 + l15) * 144 + half * 32;
      v16bf bk0 = *(const v16bf*)(krow);
      v16bf bk1 = *(const v16bf*)(krow + 64);
      v8f s = {};
      s = WMMA_BF16(aq[0], bk0, s);
      s = WMMA_BF16(aq[1], bk1, s);
      const int kcol = kb + j * 16 + l15;
#pragma unroll
      for (int v = 0; v < 8; ++v) {
        const int qr = q0 + v + 8 * half;
        s2[j][v] = (kcol <= qr) ? s[v] : -3.0e38f;  // causal mask
      }
    }
    // --- online softmax (row reductions across 16-lane halves) ---
#pragma unroll
    for (int v = 0; v < 8; ++v) {
      float mx = fmaxf(s2[0][v], s2[1][v]);
#pragma unroll
      for (int msk = 1; msk < 16; msk <<= 1) mx = fmaxf(mx, __shfl_xor(mx, msk, 32));
      const float mnew = fmaxf(mi[v], mx);
      const float scale = __expf(mi[v] - mnew);
      mi[v] = mnew;
      const float p0 = __expf(s2[0][v] - mnew);
      const float p1 = __expf(s2[1][v] - mnew);
      float rs = p0 + p1;
#pragma unroll
      for (int msk = 1; msk < 16; msk <<= 1) rs += __shfl_xor(rs, msk, 32);
      li[v] = li[v] * scale + rs;
#pragma unroll
      for (int t = 0; t < 4; ++t) acc[t][v] *= scale;
      const int r = v + 8 * half;
      *(__bf16*)(pl + r * 80 + l15 * 2)      = (__bf16)p0;
      *(__bf16*)(pl + r * 80 + 32 + l15 * 2) = (__bf16)p1;
    }
    __builtin_amdgcn_wave_barrier();
    asm volatile("" ::: "memory");  // LDS in-order per wave; block code motion

    // --- reload P in A-fragment layout ---
    v16bf pa;
    {
      const char* prow = pl + l15 * 80 + half * 32;
      v8bf lo = *(const v8bf*)(prow);
      v8bf hi = *(const v8bf*)(prow + 16);
#pragma unroll
      for (int i = 0; i < 8; ++i) { pa[i] = lo[i]; pa[8 + i] = hi[i]; }
    }
    // --- P @ V from LDS ---
#pragma unroll
    for (int t = 0; t < 4; ++t) {
      v16bf bv = *(const v16bf*)(smem + FVB(buf) + (t * 16 + l15) * 80 + half * 32);
      acc[t] = WMMA_BF16(pa, bv, acc[t]);
    }
    __builtin_amdgcn_wave_barrier();
    asm volatile("" ::: "memory");
  };

  const int kb_end  = q0b + 128;   // block-uniform causal bound
  const int kb_last = kb_end - 32;

  stage_kv(0, 0);
  for (int kb = 0; kb < kb_last; kb += 32) {
    const int buf = (kb >> 5) & 1;
    stage_kv(buf ^ 1, kb + 32);
    wait_async_le2();   // this block's 2 older ops complete (in-order)
    __syncthreads();
    if (kb < q0 + 16) fa_step(buf, kb);   // wave-uniform guard: EXEC stays ~0
    __syncthreads();
  }
  wait_async_0();
  __syncthreads();
  if (kb_last < q0 + 16) fa_step((kb_last >> 5) & 1, kb_last);

  // --- epilogue: normalize and store bf16 attn rows ---
#pragma unroll
  for (int t = 0; t < 4; ++t) {
    const int e = h * DD + t * 16 + l15;
#pragma unroll
    for (int v = 0; v < 8; ++v) {
      const int qr = q0 + v + 8 * half;
      const float o = acc[t][v] / li[v];
      O[((size_t)b * SS + qr) * EE + e] = (__bf16)o;
    }
  }
}

// ---------------------------------------------------------------------------
// Output projection: Out[4096,2048] = A[4096,2048] * BT[2048,2048]^T + bias.
// ---------------------------------------------------------------------------
__global__ __launch_bounds__(256) void out_gemm(const __bf16* __restrict__ A,
                                                const __bf16* __restrict__ BT,
                                                const float* __restrict__ bias,
                                                float* __restrict__ Out) {
  extern __shared__ char smem[];
  const int tid  = threadIdx.x;
  const int lane = tid & 31;
  const int wid  = tid >> 5;
  const int half = lane >> 4;
  const int l15  = lane & 15;
  const int m0b = blockIdx.x * 128;
  const int n0  = blockIdx.y * 64;
  const __bf16* Ablk = A + (size_t)m0b * EE;
  const __bf16* Bblk = BT + (size_t)n0 * EE;

  v8f acc[4] = {};

  auto compute_tile = [&](int buf) {
    const char* abase = smem + ABASE(buf) + (wid * 16 + l15) * LDT + half * 32;
    const char* bbase = smem + BBASE(buf) + l15 * LDT + half * 32;
    v16bf a[2], bf[2][4];
#pragma unroll
    for (int ks = 0; ks < 2; ++ks) {
      v8bf lo = *(const v8bf*)(abase + ks * 64);
      v8bf hi = *(const v8bf*)(abase + ks * 64 + 16);
#pragma unroll
      for (int i = 0; i < 8; ++i) { a[ks][i] = lo[i]; a[ks][8 + i] = hi[i]; }
#pragma unroll
      for (int t = 0; t < 4; ++t)
        bf[ks][t] = *(const v16bf*)(bbase + t * 16 * LDT + ks * 64);
    }
#pragma unroll
    for (int ks = 0; ks < 2; ++ks)
#pragma unroll
      for (int t = 0; t < 4; ++t)
        acc[t] = WMMA_BF16(a[ks], bf[ks][t], acc[t]);
  };

  load_tile_async<128, 4>(ABASE(0), Ablk, EE, tid);
  load_tile_async<64, 2>(BBASE(0), Bblk, EE, tid);

  for (int kk = 0; kk < EE - KSTEP; kk += KSTEP) {
    const int buf = (kk >> 6) & 1;
    load_tile_async<128, 4>(ABASE(buf ^ 1), Ablk + kk + KSTEP, EE, tid);
    load_tile_async<64, 2>(BBASE(buf ^ 1), Bblk + kk + KSTEP, EE, tid);
    wait_async_le6();
    __syncthreads();
    compute_tile(buf);
    __syncthreads();
  }
  wait_async_0();
  __syncthreads();
  compute_tile(((EE - KSTEP) >> 6) & 1);

#pragma unroll
  for (int t = 0; t < 4; ++t) {
    const int col = n0 + t * 16 + l15;
    const float bs = bias[col];
#pragma unroll
    for (int v = 0; v < 8; ++v) {
      const int row = m0b + wid * 16 + v + 8 * half;
      Out[(size_t)row * EE + col] = acc[t][v] + bs;
    }
  }
}

// ---------------------------------------------------------------------------
// Workspace layout (bytes, all 256B-aligned):
//   hs_bf16   [4096,2048]    @ 0          | w_qkvT [6144,2048] @ 16777216
//   w_outT    [2048,2048]    @ 41943040   | Q [B,H,S,D] @ 50331648
//   K [B,H,S,D] @ 67108864   | V [B,H,D,S] @ 83886080 | attn @ 100663296
// ---------------------------------------------------------------------------
extern "C" void kernel_launch(void* const* d_in, const int* in_sizes, int n_in,
                              void* d_out, int out_size, void* d_ws, size_t ws_size,
                              hipStream_t stream) {
  const float* hs   = (const float*)d_in[0];
  const float* wqkv = (const float*)d_in[1];
  const float* bqkv = (const float*)d_in[2];
  const float* wout = (const float*)d_in[3];
  const float* bout = (const float*)d_in[4];
  float* out = (float*)d_out;

  char* ws = (char*)d_ws;
  __bf16* hsb   = (__bf16*)(ws + 0);
  __bf16* wqkvT = (__bf16*)(ws + 16777216);
  __bf16* woutT = (__bf16*)(ws + 41943040);
  __bf16* Qb    = (__bf16*)(ws + 50331648);
  __bf16* Kb    = (__bf16*)(ws + 67108864);
  __bf16* Vb    = (__bf16*)(ws + 83886080);
  __bf16* Ab    = (__bf16*)(ws + 100663296);

  // 1) precision / layout prep
  cvt_bf16<<<(MM * EE + 255) / 256, 256, 0, stream>>>(hs, hsb, MM * EE);
  cvt_t_bf16<<<dim3(EE / 32, N_QKV / 32), 256, 0, stream>>>(wqkv, wqkvT, EE, N_QKV);
  cvt_t_bf16<<<dim3(EE / 32, EE / 32), 256, 0, stream>>>(wout, woutT, EE, EE);

  // 2) QKV projection (M=4096 x N=6144 x K=2048), async-LDS double buffered
  qkv_gemm<<<dim3(MM / 128, N_QKV / 64), 256, GEMM_SMEM, stream>>>(hsb, wqkvT, bqkv,
                                                                   Qb, Kb, Vb);

  // 3) causal flash attention, block-shared async K/V staging
  flash_attn<<<dim3(BB * HH, SS / 128), 256, FLASH_SMEM, stream>>>(Qb, Kb, Vb, Ab);

  // 4) output projection (M=4096 x N=2048 x K=2048), fp32 out
  out_gemm<<<dim3(MM / 128, EE / 64), 256, GEMM_SMEM, stream>>>(Ab, woutT, bout, out);
}
